// BondAngleGraphEmbedding_34076270526996
// MI455X (gfx1250) — matrix operand order repaired and checked by reading, not compiled, not run on hardware
//
#include <hip/hip_runtime.h>
#include <hip/hip_bf16.h>

typedef __attribute__((ext_vector_type(16))) _Float16 v16h;
typedef __attribute__((ext_vector_type(8)))  float    v8f;
typedef __attribute__((ext_vector_type(4)))  float    v4f;

#define N_NODES 1536
#define N_EDGES 49152
#define DIM 64

// ---------------------------------------------------------------------------
// angle_emb[e, :] = RBF(angle[e], 32 centers) @ W_ang + b_ang   via WMMA f16
// One wave handles 16 rows; 4 WMMAs cover the 64 output dims.
// ---------------------------------------------------------------------------
__global__ void angle_emb_wmma_kernel(const float* __restrict__ angle,
                                      const float* __restrict__ W_ang,
                                      const float* __restrict__ b_ang,
                                      float* __restrict__ emb) {
  const int lane = threadIdx.x & 31;
  const int wave = (int)((blockIdx.x * (long long)blockDim.x + threadIdx.x) >> 5);
  const int row0 = wave * 16;
  if (row0 >= N_EDGES) return;                  // wave-uniform, never taken
  const int m  = lane & 15;                     // row within tile / col within N-tile
  const int hs = lane >> 4;                     // half-wave select

  // A-matrix (16x32 f16): lanes 0-15 hold K{0..7,16..23}, lanes 16-31 K{8..15,24..31}
  const float a = angle[row0 + m];
  v16h A;
#pragma unroll
  for (int i = 0; i < 16; ++i) {
    const int k = (i < 8) ? (hs * 8 + i) : (hs * 8 + 8 + i);
    const float d = a - 0.1f * (float)k;
    A[i] = (_Float16)__expf(-10.0f * d * d);
  }

#pragma unroll
  for (int nt = 0; nt < 4; ++nt) {
    const int n = nt * 16 + m;                  // output dim
    // B-matrix (32x16 f16): lanes 0-15 hold K{0..7,16..23}, lanes 16-31 K{8..15,24..31}
    v16h B;
#pragma unroll
    for (int i = 0; i < 16; ++i) {
      const int k = (i < 8) ? (hs * 8 + i) : (hs * 8 + 8 + i);
      B[i] = (_Float16)W_ang[k * DIM + n];
    }
    const float bias = b_ang[n];
    v8f c;
#pragma unroll
    for (int r = 0; r < 8; ++r) c[r] = bias;    // bias depends only on N(=lane)
    c = __builtin_amdgcn_wmma_f32_16x16x32_f16(false, A, false, B,
                                               (short)0, c, false, false);
#pragma unroll
    for (int r = 0; r < 8; ++r) {
      const int mrow = hs * 8 + r;              // C layout: VGPR r -> M = r + 8*hs
      emb[(size_t)(row0 + mrow) * DIM + n] = c[r];
    }
  }
}

// ---------------------------------------------------------------------------
// bond_feat[v,:] = E_dir[..] + E_type[..] + E_ring[..] + RBF(len,20)@W_len + b
// K padded 20 -> 32 with zeros (compile-time predicate in unrolled fill).
// ---------------------------------------------------------------------------
__global__ void bond_feat_wmma_kernel(const int* __restrict__ bond_dir,
                                      const int* __restrict__ bond_type,
                                      const int* __restrict__ is_in_ring,
                                      const float* __restrict__ length,
                                      const float* __restrict__ E_dir,
                                      const float* __restrict__ E_type,
                                      const float* __restrict__ E_ring,
                                      const float* __restrict__ W_len,
                                      const float* __restrict__ b_len,
                                      float* __restrict__ out) {
  const int lane = threadIdx.x & 31;
  const int wave = (int)((blockIdx.x * (long long)blockDim.x + threadIdx.x) >> 5);
  const int row0 = wave * 16;
  if (row0 >= N_NODES) return;                  // wave-uniform, never taken
  const int m  = lane & 15;
  const int hs = lane >> 4;

  const float x = length[row0 + m];
  v16h A;
#pragma unroll
  for (int i = 0; i < 16; ++i) {
    const int k = (i < 8) ? (hs * 8 + i) : (hs * 8 + 8 + i);
    float v = 0.0f;
    if (k < 20) {                               // compile-time after unroll
      const float d = x - 0.1f * (float)k;
      v = __expf(-10.0f * d * d);
    }
    A[i] = (_Float16)v;
  }

#pragma unroll
  for (int nt = 0; nt < 4; ++nt) {
    const int n = nt * 16 + m;
    v16h B;
#pragma unroll
    for (int i = 0; i < 16; ++i) {
      const int k = (i < 8) ? (hs * 8 + i) : (hs * 8 + 8 + i);
      B[i] = (_Float16)((k < 20) ? W_len[k * DIM + n] : 0.0f);
    }
    const float bias = b_len[n];
    v8f c;
#pragma unroll
    for (int r = 0; r < 8; ++r) c[r] = bias;
    c = __builtin_amdgcn_wmma_f32_16x16x32_f16(false, A, false, B,
                                               (short)0, c, false, false);
#pragma unroll
    for (int r = 0; r < 8; ++r) {
      const int node = row0 + hs * 8 + r;
      const float e = E_dir [bond_dir [node] * DIM + n]
                    + E_type[bond_type[node] * DIM + n]
                    + E_ring[is_in_ring[node] * DIM + n];
      out[(size_t)node * DIM + n] = c[r] + e;
    }
  }
}

// ---------------------------------------------------------------------------
// angle_matrix[i,j,:] = emb[map[i,j],:]  — the 604 MB streaming gather.
// 16 threads per 256B row (float4 each): fully coalesced loads (L2-resident
// emb table) and contiguous non-temporal stores (output >> L2, don't cache).
// ---------------------------------------------------------------------------
__global__ void gather_kernel(const int* __restrict__ map,
                              const float* __restrict__ emb,
                              float* __restrict__ out) {
  const long long tid   = (long long)blockIdx.x * blockDim.x + threadIdx.x;
  const long long entry = tid >> 4;
  const int       q     = (int)(tid & 15);
  const int idx = map[entry];
  const v4f* __restrict__ src = (const v4f*)(emb + (size_t)idx * DIM);
  const v4f v = src[q];
  v4f* dst = (v4f*)(out + (size_t)entry * DIM);
  __builtin_nontemporal_store(v, dst + q);
}

// edge_index passthrough (int -> float in concatenated output buffer)
__global__ void edge_copy_kernel(const int* __restrict__ e,
                                 float* __restrict__ out) {
  const int i = blockIdx.x * blockDim.x + threadIdx.x;
  if (i < 2 * N_EDGES) out[i] = (float)e[i];
}

extern "C" void kernel_launch(void* const* d_in, const int* in_sizes, int n_in,
                              void* d_out, int out_size, void* d_ws, size_t ws_size,
                              hipStream_t stream) {
  const int*   bond_dir   = (const int*)  d_in[0];
  const int*   bond_type  = (const int*)  d_in[1];
  const int*   is_in_ring = (const int*)  d_in[2];
  const float* length     = (const float*)d_in[3];
  const float* angle      = (const float*)d_in[4];
  const int*   edge_index = (const int*)  d_in[5];
  const int*   nmap       = (const int*)  d_in[6];
  const float* E_dir      = (const float*)d_in[7];
  const float* E_type     = (const float*)d_in[8];
  const float* E_ring     = (const float*)d_in[9];
  const float* W_len      = (const float*)d_in[10];
  const float* b_len      = (const float*)d_in[11];
  const float* W_ang      = (const float*)d_in[12];
  const float* b_ang      = (const float*)d_in[13];

  float* out            = (float*)d_out;
  float* bond_feat_out  = out;                                            // 1536*64
  float* angle_mat_out  = out + (size_t)N_NODES * DIM;                    // 1536*1536*64
  float* edge_out       = angle_mat_out + (size_t)N_NODES * N_NODES * DIM;// 2*49152
  float* emb_ws         = (float*)d_ws;                                   // 49152*64 f32

  // angle_emb: 49152/16 = 3072 waves -> 8 waves/block -> 384 blocks
  angle_emb_wmma_kernel<<<(N_EDGES / 16) * 32 / 256, 256, 0, stream>>>(
      angle, W_ang, b_ang, emb_ws);

  // bond_feat: 1536/16 = 96 waves -> 12 blocks
  bond_feat_wmma_kernel<<<(N_NODES / 16) * 32 / 256, 256, 0, stream>>>(
      bond_dir, bond_type, is_in_ring, length,
      E_dir, E_type, E_ring, W_len, b_len, bond_feat_out);

  // gather: 1536*1536 entries * 16 threads = 37,748,736 threads -> 147,456 blocks
  const long long gthreads = (long long)N_NODES * N_NODES * 16;
  gather_kernel<<<(unsigned)(gthreads / 256), 256, 0, stream>>>(
      nmap, emb_ws, angle_mat_out);

  edge_copy_kernel<<<(2 * N_EDGES + 255) / 256, 256, 0, stream>>>(
      edge_index, edge_out);
}
